// LayerGate_36919538876504
// MI455X (gfx1250) — matrix operand order repaired and checked by reading, not compiled, not run on hardware
//
#include <hip/hip_runtime.h>

typedef __attribute__((ext_vector_type(16))) _Float16 v16h;
typedef __attribute__((ext_vector_type(8)))  _Float16 v8h;
typedef __attribute__((ext_vector_type(2)))  __fp16   h2raw;   // return type of cvt_pkrtz
typedef __attribute__((ext_vector_type(8)))  float    v8f;
typedef __attribute__((ext_vector_type(4)))  float    f32x4;
typedef __attribute__((ext_vector_type(2)))  unsigned int u32x2;

#define Hc   128
#define Sc   512
#define Bc   32
#define TM   64          // rows (s) per workgroup
#define KCAT 384
#define KEXP 128

// LDS layout in halves; +8-half pads keep b128 fragment loads conflict-free.
// B regions first: the 48 B_cat fragment loads then address LDS below 64KB,
// reachable with the 16-bit DS immediate offset (no v_add address math).
#define LDA_CAT 392
#define LDA_EXP 136
#define LDB_CAT 72
#define LDB_EXP 56
#define OFF_BCAT 0                               // 0       .. 55296 B
#define OFF_BEXP (KCAT * LDB_CAT)                // 27648 h  (55296 B)
#define OFF_ACAT (OFF_BEXP + KEXP * LDB_EXP)     // 34816 h  (69632 B)
#define OFF_AEXP (OFF_ACAT + TM * LDA_CAT)       // 59904 h  (119808 B)
#define SMEM_HALVES (OFF_AEXP + TM * LDA_EXP)    // 68608 halves = 137216 B

// f32x4 -> 4 halves using packed converts (v_cvt_pkrtz_f16_f32), 8-byte LDS store
static __device__ __forceinline__ void st4h(_Float16* d, f32x4 v) {
  h2raw lo = __builtin_amdgcn_cvt_pkrtz(v.x, v.y);
  h2raw hi = __builtin_amdgcn_cvt_pkrtz(v.z, v.w);
  u32x2 w;
  w.x = __builtin_bit_cast(unsigned int, lo);
  w.y = __builtin_bit_cast(unsigned int, hi);
  *(u32x2*)d = w;   // ds_store_b64
}

// Build a v16h fragment: elements 0..7 from p0, 8..15 from p1 (two ds_load_b128)
static __device__ __forceinline__ v16h ld_frag(const _Float16* p0, const _Float16* p1) {
  v8h a = *(const v8h*)p0;
  v8h b = *(const v8h*)p1;
  v16h r;
#pragma unroll
  for (int i = 0; i < 8; ++i) { r[i] = a[i]; r[i + 8] = b[i]; }
  return r;
}

// fast sigmoid via native TRANS ops (v_exp_f32, v_rcp_f32)
static __device__ __forceinline__ float fsig(float x) {
  return __builtin_amdgcn_rcpf(1.0f + __expf(-x));
}

__global__ __launch_bounds__(128)
void layergate_fused(const float* __restrict__ cnn,   // (32, 2048, 128)
                     const float* __restrict__ gaz,   // (32, 512, 128)
                     const float* __restrict__ gazb,  // (32, 512, 128)
                     const float* __restrict__ gm,    // (32, 512, 128)
                     const float* __restrict__ expin, // (32, 512, 128)
                     const float* __restrict__ Wcat,  // (384, 2048)
                     const float* __restrict__ bcat,  // (2048)
                     const float* __restrict__ Wexp,  // (128, 1536)
                     const float* __restrict__ bexp,  // (1536)
                     float* __restrict__ out)         // (4, 32, 512, 128) tuple-flat
{
  extern __shared__ _Float16 sm[];
  const int tid = threadIdx.x;
  const int bid = blockIdx.x;
  const int hg  = bid & 7;          // h-group (fastest -> A-tile L2 reuse)
  const int rt  = (bid >> 3) & 7;   // row tile within chunk
  const int bj  = bid >> 6;         // 0..127
  const int j   = bj & 3;           // sequence chunk
  const int b   = bj >> 2;          // batch
  const int m0  = rt * TM;
  const int h0  = hg * 16;

  const float* gsrc = (j & 1) ? gazb : gaz;   // concat([g, gb, g, gb], axis=1)

  // ---------------- stage A_cat: rows = [CNN | gaz_sel | gm], K = 384 ----------------
  {
    const float* p0 = cnn  + ((size_t)(b * 2048 + j * Sc + m0)) * Hc;
    const float* p1 = gsrc + ((size_t)(b * Sc + m0)) * Hc;
    const float* p2 = gm   + ((size_t)(b * Sc + m0)) * Hc;
#pragma unroll 2
    for (int idx = tid; idx < TM * 32; idx += 128) {
      int r = idx >> 5, kq = (idx & 31) * 4;
      st4h(&sm[OFF_ACAT + r * LDA_CAT + kq], *(const f32x4*)(p0 + (size_t)r * Hc + kq));
    }
#pragma unroll 2
    for (int idx = tid; idx < TM * 32; idx += 128) {
      int r = idx >> 5, kq = (idx & 31) * 4;
      st4h(&sm[OFF_ACAT + r * LDA_CAT + 128 + kq], *(const f32x4*)(p1 + (size_t)r * Hc + kq));
    }
#pragma unroll 2
    for (int idx = tid; idx < TM * 32; idx += 128) {
      int r = idx >> 5, kq = (idx & 31) * 4;
      st4h(&sm[OFF_ACAT + r * LDA_CAT + 256 + kq], *(const f32x4*)(p2 + (size_t)r * Hc + kq));
    }
    // A_exp: exper_input rows, K = 128
    const float* pe = expin + ((size_t)(b * Sc + m0)) * Hc;
#pragma unroll 2
    for (int idx = tid; idx < TM * 32; idx += 128) {
      int r = idx >> 5, kq = (idx & 31) * 4;
      st4h(&sm[OFF_AEXP + r * LDA_EXP + kq], *(const f32x4*)(pe + (size_t)r * Hc + kq));
    }
  }
  // ---------------- stage B_cat: K=384 x 64 cols {j*512 + c*128 + h0 + 0..15} ----------------
  {
#pragma unroll 2
    for (int idx = tid; idx < KCAT * 16; idx += 128) {
      int k = idx >> 4, rem = idx & 15;
      int c = rem >> 2, q = (rem & 3) * 4;
      f32x4 v = *(const f32x4*)(Wcat + (size_t)k * 2048 + j * 512 + c * 128 + h0 + q);
      st4h(&sm[OFF_BCAT + k * LDB_CAT + c * 16 + q], v);
    }
    // B_exp: K=128 x 48 cols {j*384 + c*128 + h0 + 0..15}
#pragma unroll 2
    for (int idx = tid; idx < KEXP * 12; idx += 128) {
      int k = idx / 12, rem = idx % 12;
      int c = rem >> 2, q = (rem & 3) * 4;
      f32x4 v = *(const f32x4*)(Wexp + (size_t)k * 1536 + j * 384 + c * 128 + h0 + q);
      st4h(&sm[OFF_BEXP + k * LDB_EXP + c * 16 + q], v);
    }
  }
  __syncthreads();

  // ---------------- WMMA: each wave owns a 16-row tile, 4 cat + 3 exp col-tiles ----------------
  const int lane = tid & 31;
  const int wave = tid >> 5;
  const int wrow = wave * 16;
  const int lrow = lane & 15;
  const int lhi  = lane >> 4;

  v8f aC0 = {}, aC1 = {}, aC2 = {}, aC3 = {}, aE0 = {}, aE1 = {}, aE2 = {};

  // A frag: lane = local row; halves 0..7 = K[k0+8*lhi .. +7], 8..15 = K[k0+8*lhi+16 ..]
  const _Float16* aB  = &sm[OFF_ACAT + (wrow + lrow) * LDA_CAT + lhi * 8];
  const _Float16* aeB = &sm[OFF_AEXP + (wrow + lrow) * LDA_EXP + lhi * 8];
  // B frag: lane holds N-col lrow; halves 0..7 = K[k0+8*lhi .. +7], 8..15 = K[k0+8*lhi+16 ..]
  const _Float16* bB  = &sm[OFF_BCAT + (lhi * 8) * LDB_CAT + lrow];
  const _Float16* beB = &sm[OFF_BEXP + (lhi * 8) * LDB_EXP + lrow];

  // Fused K loop: exp GEMM folded into the first 4 cat iterations for ILP
  // (7 independent WMMAs + 9 fragment loads per early iteration -> fills
  //  WMMA->WMMA co-execution hazard slots instead of v_nop padding)
#pragma unroll
  for (int kt = 0; kt < 12; ++kt) {
    const int k0 = kt * 32;
    v16h af = ld_frag(aB + k0, aB + k0 + 16);
    const _Float16* bp = bB + k0 * LDB_CAT;
    v16h b0, b1, b2, b3;
#pragma unroll
    for (int i = 0; i < 8; ++i) {
      b0[i] = bp[i * LDB_CAT +  0]; b0[i + 8] = bp[(16 + i) * LDB_CAT +  0];
      b1[i] = bp[i * LDB_CAT + 16]; b1[i + 8] = bp[(16 + i) * LDB_CAT + 16];
      b2[i] = bp[i * LDB_CAT + 32]; b2[i + 8] = bp[(16 + i) * LDB_CAT + 32];
      b3[i] = bp[i * LDB_CAT + 48]; b3[i + 8] = bp[(16 + i) * LDB_CAT + 48];
    }
    aC0 = __builtin_amdgcn_wmma_f32_16x16x32_f16(false, af, false, b0, (short)0, aC0, false, false);
    aC1 = __builtin_amdgcn_wmma_f32_16x16x32_f16(false, af, false, b1, (short)0, aC1, false, false);
    aC2 = __builtin_amdgcn_wmma_f32_16x16x32_f16(false, af, false, b2, (short)0, aC2, false, false);
    aC3 = __builtin_amdgcn_wmma_f32_16x16x32_f16(false, af, false, b3, (short)0, aC3, false, false);
    if (kt < 4) {   // statically resolved under full unroll
      v16h ae = ld_frag(aeB + k0, aeB + k0 + 16);
      const _Float16* ep = beB + k0 * LDB_EXP;
      v16h e0, e1, e2;
#pragma unroll
      for (int i = 0; i < 8; ++i) {
        e0[i] = ep[i * LDB_EXP +  0]; e0[i + 8] = ep[(16 + i) * LDB_EXP +  0];
        e1[i] = ep[i * LDB_EXP + 16]; e1[i + 8] = ep[(16 + i) * LDB_EXP + 16];
        e2[i] = ep[i * LDB_EXP + 32]; e2[i + 8] = ep[(16 + i) * LDB_EXP + 32];
      }
      aE0 = __builtin_amdgcn_wmma_f32_16x16x32_f16(false, ae, false, e0, (short)0, aE0, false, false);
      aE1 = __builtin_amdgcn_wmma_f32_16x16x32_f16(false, ae, false, e1, (short)0, aE1, false, false);
      aE2 = __builtin_amdgcn_wmma_f32_16x16x32_f16(false, ae, false, e2, (short)0, aE2, false, false);
    }
  }

  // ---------------- epilogue: tanh / sigmoid / softmax / blend (native TRANS) ----------------
  // C/D layout: element i -> row M = i + 8*lhi (within wave tile), col N = lrow
  const int h = h0 + lrow;
  const float bc0 = bcat[j * 512 +       h];
  const float bc1 = bcat[j * 512 + 128 + h];
  const float bc2 = bcat[j * 512 + 256 + h];
  const float bc3 = bcat[j * 512 + 384 + h];
  const float be0 = bexp[j * 384 +       h];
  const float be1 = bexp[j * 384 + 128 + h];
  const float be2 = bexp[j * 384 + 256 + h];

  const int srow = m0 + wrow + lhi * 8;   // s index of element i is srow + i
  const float* cnnR = cnn   + ((size_t)(b * 2048 + j * Sc + srow)) * Hc + h;
  const float* expR = expin + ((size_t)(b * Sc + srow)) * Hc + h;
  // tuple-flat output: chunk-major (j, b, s, h)
  float* outR = out + (((size_t)j * Bc * Sc) + (size_t)b * Sc + srow) * Hc + h;

#pragma unroll
  for (int i = 0; i < 8; ++i) {
    // tanh(x) = (e^{2x} - 1) / (e^{2x} + 1), clamped so e^{2x} stays finite
    float xt = aC0[i] + bc0;
    xt = fminf(fmaxf(xt, -15.0f), 15.0f);
    float t  = __expf(2.0f * xt);
    float ns = (t - 1.0f) * __builtin_amdgcn_rcpf(t + 1.0f);

    float s0 = fsig(aC1[i] + bc1 + aE0[i] + be0);
    float s1 = fsig(aC2[i] + bc2 + aE1[i] + be1);
    float s2 = fsig(aC3[i] + bc3 + aE2[i] + be2);
    float e0 = __expf(s0), e1 = __expf(s1), e2 = __expf(s2);
    float inv = __builtin_amdgcn_rcpf(e0 + e1 + e2);
    float cv = cnnR[(size_t)i * Hc];
    float ev = expR[(size_t)i * Hc];
    outR[(size_t)i * Hc] = (e0 * ns + e1 * cv + e2 * ev) * inv;
  }
}

extern "C" void kernel_launch(void* const* d_in, const int* in_sizes, int n_in,
                              void* d_out, int out_size, void* d_ws, size_t ws_size,
                              hipStream_t stream) {
  (void)in_sizes; (void)n_in; (void)out_size; (void)d_ws; (void)ws_size;
  const float* cnn   = (const float*)d_in[0];
  const float* gaz   = (const float*)d_in[1];
  const float* gazb  = (const float*)d_in[2];
  const float* gm    = (const float*)d_in[3];
  const float* expin = (const float*)d_in[4];
  const float* Wcat  = (const float*)d_in[5];
  const float* bcat  = (const float*)d_in[6];
  const float* Wexp  = (const float*)d_in[7];
  const float* bexp  = (const float*)d_in[8];
  float* out = (float*)d_out;

  const size_t shmem = (size_t)SMEM_HALVES * sizeof(_Float16); // 137216 B (<320KB/WGP)
  static bool attr_set = false;  // host-side, idempotent, not a stream op
  if (!attr_set) {
    (void)hipFuncSetAttribute((const void*)layergate_fused,
                              hipFuncAttributeMaxDynamicSharedMemorySize, (int)shmem);
    attr_set = true;
  }
  // grid: 32 batches * 4 chunks * 8 row-tiles * 8 h-groups = 8192 blocks, 4 waves each
  layergate_fused<<<dim3(8192), dim3(128), shmem, stream>>>(
      cnn, gaz, gazb, gm, expin, Wcat, bcat, Wexp, bexp, out);
}
